// HC2MMoEModel_86955907875130
// MI455X (gfx1250) — matrix operand order repaired and checked by reading, not compiled, not run on hardware
//
#include <hip/hip_runtime.h>

// ---------------- problem constants ----------------
#define B_   16384
#define IN_  1024
#define E_   6
#define D_   20
#define EH1_ 256
#define EH2_ 128
#define EO_  10
#define GH_  64
#define TH_  64

typedef unsigned short u16;
typedef __attribute__((ext_vector_type(16))) __bf16 v16bf;
typedef __attribute__((ext_vector_type(8)))  float  v8f;

union Frag {
    uint4 u2[2];
    v16bf v;
};

#define WMMA_BF16(a, b, c) \
    __builtin_amdgcn_wmma_f32_16x16x32_bf16(false, (a), false, (b), (short)0, (c), false, false)

static __device__ __forceinline__ u16 f2bf(float f) {
    union { float f; unsigned u; } v; v.f = f;
    unsigned r = v.u + 0x7FFFu + ((v.u >> 16) & 1u);   // RNE
    return (u16)(r >> 16);
}
static __device__ __forceinline__ float bf2f(u16 h) {
    union { unsigned u; float f; } v; v.u = ((unsigned)h) << 16;
    return v.f;
}

// ---------------- conversion kernels ----------------
__global__ void cvt_f32_bf16(const float* __restrict__ in, u16* __restrict__ out, int n) {
    int i = blockIdx.x * blockDim.x + threadIdx.x;
    if (i < n) out[i] = f2bf(in[i]);
}

// in: [mats][K][N] f32 (row major)  ->  out: [mats][N][K] bf16 (column major per matrix)
__global__ void cvt_transpose_bf16(const float* __restrict__ in, u16* __restrict__ out,
                                   int mats, int K, int N) {
    int i = blockIdx.x * blockDim.x + threadIdx.x;
    int total = mats * K * N;
    if (i >= total) return;
    int m = i / (K * N);
    int r = i % (K * N);
    int k = r / N;
    int n = r % N;
    out[((size_t)m * N + n) * K + k] = f2bf(in[i]);
}

// Ew3 [E][128][10] f32 -> Ew3T [E][16][128] bf16, rows 10..15 zero-padded
__global__ void cvt_ew3(const float* __restrict__ in, u16* __restrict__ out) {
    int i = blockIdx.x * blockDim.x + threadIdx.x;
    if (i >= E_ * 16 * 128) return;
    int e = i >> 11;
    int r = i & 2047;
    int n = r >> 7;      // output row (= EO column), 0..15
    int k = r & 127;     // K index
    float v = (n < EO_) ? in[((size_t)e * 128 + k) * EO_ + n] : 0.f;
    out[i] = f2bf(v);
}

// ---------------- domain bucketing ----------------
__global__ void bucket_init(int* counts, int* cursor) {
    int i = threadIdx.x;
    if (i < D_) { counts[i] = 0; cursor[i] = 0; }
}
__global__ void bucket_count(const int* __restrict__ dom, int* counts, int n) {
    int i = blockIdx.x * blockDim.x + threadIdx.x;
    if (i < n) atomicAdd(&counts[dom[i]], 1);
}
__global__ void bucket_scan(const int* __restrict__ counts, int* offs) {
    if (threadIdx.x == 0) {
        int s = 0;
        for (int d = 0; d < D_; ++d) { offs[d] = s; s += counts[d]; }
    }
}
__global__ void bucket_scatter(const int* __restrict__ dom, const int* __restrict__ offs,
                               int* cursor, int* idxlist, int n) {
    int i = blockIdx.x * blockDim.x + threadIdx.x;
    if (i < n) {
        int d = dom[i];
        int p = atomicAdd(&cursor[d], 1);
        idxlist[offs[d] + p] = i;
    }
}

// ---------------- main expert GEMMs (bf16 WMMA, f32 acc, fused bias+ReLU) ----------------
// 8 waves per block (4 M x 2 N), each wave computes a 32x64 tile as 2x4 WMMA subtiles:
// 8 WMMAs per 12 b128 loads per K-step, only ~120 hot VGPRs (no shuffle pressure).
// A: [(e)][M][K] bf16 row-major (aExpStride elements between experts; 0 = shared)
// BT: [E][NTOT][K] bf16 (column major weights)
// C: [E][M][NTOT] bf16 = relu(A*B + bias)
template <int K, int NTOT>
__global__ __launch_bounds__(256) void gemm_bias_relu_bf16(
    const u16* __restrict__ A, long long aExpStride,
    const u16* __restrict__ BT, const float* __restrict__ bias,
    u16* __restrict__ C, int M)
{
    int e    = blockIdx.z;
    int tid  = threadIdx.x;
    int lane = tid & 31;
    int w    = tid >> 5;       // 8 waves: 4 (M) x 2 (N)
    int wm   = w >> 1;
    int wn   = w & 1;
    int m0   = blockIdx.x * 128 + wm * 32;
    int n0   = blockIdx.y * 128 + wn * 64;
    int lrow  = lane & 15;
    int lhalf = lane >> 4;

    const u16* Ae = A + (size_t)e * (size_t)aExpStride;
    const u16* aRow[2];
    const u16* bCol[4];
    #pragma unroll
    for (int i = 0; i < 2; ++i)
        aRow[i] = Ae + (size_t)(m0 + i * 16 + lrow) * K;
    #pragma unroll
    for (int j = 0; j < 4; ++j)
        bCol[j] = BT + ((size_t)e * NTOT + n0 + j * 16 + lrow) * K;

    v8f acc[2][4] = {};

    #pragma unroll 2
    for (int k0 = 0; k0 < K; k0 += 32) {
        Frag a[2], b[4];
        int ka = k0 + lhalf * 16;  // A layout: lanes<16: K 0..15 ; lanes>=16: K 16..31
        int kb = k0 + lhalf * 16;  // B layout: lanes<16: K 0..15 of col ; lanes>=16: K 16..31
        #pragma unroll
        for (int i = 0; i < 2; ++i) {
            a[i].u2[0] = *(const uint4*)(aRow[i] + ka);
            a[i].u2[1] = *(const uint4*)(aRow[i] + ka + 8);
        }
        #pragma unroll
        for (int j = 0; j < 4; ++j) {
            b[j].u2[0] = *(const uint4*)(bCol[j] + kb);
            b[j].u2[1] = *(const uint4*)(bCol[j] + kb + 8);
        }
        #pragma unroll
        for (int i = 0; i < 2; ++i)
            #pragma unroll
            for (int j = 0; j < 4; ++j)
                acc[i][j] = WMMA_BF16(a[i].v, b[j].v, acc[i][j]);
    }

    const float* be = bias + e * NTOT;
    #pragma unroll
    for (int i = 0; i < 2; ++i) {
        #pragma unroll
        for (int j = 0; j < 4; ++j) {
            int col = n0 + j * 16 + lrow;
            float bv = be[col];
            size_t base = ((size_t)e * M + m0 + i * 16 + lhalf * 8) * NTOT + col;
            #pragma unroll
            for (int r = 0; r < 8; ++r) {
                float v = fmaxf(acc[i][j][r] + bv, 0.f);
                C[base + (size_t)r * NTOT] = f2bf(v);
            }
        }
    }
}

// expert layer 3: [M,128] x [128,16(pad)] -> eo f32 [E][M][16]  (bias, no act)
__global__ __launch_bounds__(128) void gemm3_eo(
    const u16* __restrict__ h2, const u16* __restrict__ Ew3T,
    const float* __restrict__ Eb3, float* __restrict__ eo, int M)
{
    int e    = blockIdx.y;
    int tid  = threadIdx.x;
    int lane = tid & 31;
    int w    = tid >> 5;
    int m0   = (blockIdx.x * 4 + w) * 16;
    int lrow  = lane & 15;
    int lhalf = lane >> 4;

    const u16* aRow = h2 + ((size_t)e * M + m0 + lrow) * 128;
    const u16* bCol = Ew3T + ((size_t)e * 16 + lrow) * 128;

    v8f c = {};
    #pragma unroll
    for (int k0 = 0; k0 < 128; k0 += 32) {
        Frag a, b;
        a.u2[0] = *(const uint4*)(aRow + k0 + lhalf * 16);
        a.u2[1] = *(const uint4*)(aRow + k0 + lhalf * 16 + 8);
        b.u2[0] = *(const uint4*)(bCol + k0 + lhalf * 16);
        b.u2[1] = *(const uint4*)(bCol + k0 + lhalf * 16 + 8);
        c = WMMA_BF16(a.v, b.v, c);
    }
    int col = lrow;
    float bv = (col < EO_) ? Eb3[e * EO_ + col] : 0.f;
    size_t base = ((size_t)e * M + m0 + lhalf * 8) * 16 + col;
    #pragma unroll
    for (int r = 0; r < 8; ++r) eo[base + (size_t)r * 16] = c[r] + bv;
}

// ---------------- grouped gate GEMM (only each sample's own domain) ----------------
// One wave per block; each tile is 16 gathered rows x all 64 gate-hidden cols:
// per K-step: 1 A frag reused across 4 B frags -> 4 WMMAs / 10 b128 loads.
__global__ __launch_bounds__(32) void gate_gemm(
    const u16* __restrict__ xb, const u16* __restrict__ Gw1T, const float* __restrict__ Gb1,
    const int* __restrict__ idxlist, const int* __restrict__ offs, const int* __restrict__ counts,
    u16* __restrict__ ghid)
{
    int d = blockIdx.x;
    int cnt = counts[d];
    if (cnt == 0) return;
    int off = offs[d];
    int ntiles = (cnt + 15) >> 4;

    int tid  = threadIdx.x;
    int lane = tid & 31;
    int lrow  = lane & 15;
    int lhalf = lane >> 4;

    __shared__ int sIds[16];

    for (int t = blockIdx.y; t < ntiles; t += gridDim.y) {
        __syncthreads();           // protect sIds WAR across iterations
        if (tid < 16) {
            int j = t * 16 + tid;
            sIds[tid] = idxlist[off + (j < cnt ? j : 0)];   // clamp-pad with row 0 (benign dup)
        }
        __syncthreads();

        const u16* aRow = xb + (size_t)sIds[lrow] * IN_;
        const u16* bCol[4];
        #pragma unroll
        for (int j = 0; j < 4; ++j)
            bCol[j] = Gw1T + ((size_t)d * GH_ + j * 16 + lrow) * IN_;

        v8f acc[4] = {};
        #pragma unroll 2
        for (int k0 = 0; k0 < IN_; k0 += 32) {
            Frag a, b[4];
            int ka = k0 + lhalf * 16;
            a.u2[0] = *(const uint4*)(aRow + ka);
            a.u2[1] = *(const uint4*)(aRow + ka + 8);
            int kb = k0 + lhalf * 16;
            #pragma unroll
            for (int j = 0; j < 4; ++j) {
                b[j].u2[0] = *(const uint4*)(bCol[j] + kb);
                b[j].u2[1] = *(const uint4*)(bCol[j] + kb + 8);
            }
            #pragma unroll
            for (int j = 0; j < 4; ++j)
                acc[j] = WMMA_BF16(a.v, b[j].v, acc[j]);
        }

        #pragma unroll
        for (int j = 0; j < 4; ++j) {
            int col = j * 16 + lrow;
            float bv = Gb1[d * GH_ + col];
            #pragma unroll
            for (int r = 0; r < 8; ++r) {
                int rr = lhalf * 8 + r;
                if (t * 16 + rr < cnt) {
                    int s = sIds[rr];
                    ghid[(size_t)s * GH_ + col] = f2bf(fmaxf(acc[j][r] + bv, 0.f));
                }
            }
        }
    }
}

// ---------------- fused per-sample tail: gate L2 + softmax + mmoe + tower + outputs ----------------
__global__ __launch_bounds__(256) void finalize(
    const int* __restrict__ dom, const u16* __restrict__ ghid, const float* __restrict__ eo,
    const float* __restrict__ Gw2, const float* __restrict__ Gb2,
    const float* __restrict__ Tw1, const float* __restrict__ Tb1,
    const float* __restrict__ Tw2, const float* __restrict__ Tb2,
    float* __restrict__ out)
{
    int b = blockIdx.x * blockDim.x + threadIdx.x;
    if (b >= B_) return;
    int d = dom[b];

    float gh[GH_];
    #pragma unroll 8
    for (int h = 0; h < GH_; ++h) gh[h] = bf2f(ghid[(size_t)b * GH_ + h]);

    // gate logits + softmax over E=6
    float logit[E_];
    float mx = -1e30f;
    #pragma unroll
    for (int e = 0; e < E_; ++e) {
        float a = Gb2[d * E_ + e];
        for (int h = 0; h < GH_; ++h) a += gh[h] * Gw2[(d * GH_ + h) * E_ + e];
        logit[e] = a;
        mx = fmaxf(mx, a);
    }
    float denom = 0.f, gate[E_];
    #pragma unroll
    for (int e = 0; e < E_; ++e) { gate[e] = expf(logit[e] - mx); denom += gate[e]; }
    float inv = 1.f / denom;
    #pragma unroll
    for (int e = 0; e < E_; ++e) gate[e] *= inv;

    // mmoe combine (own domain) + expert mean
    float mmoe[EO_], avg[EO_];
    #pragma unroll
    for (int o = 0; o < EO_; ++o) {
        float am = 0.f, aa = 0.f;
        #pragma unroll
        for (int e = 0; e < E_; ++e) {
            float v = eo[((size_t)e * B_ + b) * 16 + o];
            am += gate[e] * v;
            aa += v;
        }
        mmoe[o] = am;
        avg[o]  = aa * (1.f / E_);
    }

    // tower (own domain): EO->64 relu, 64->1 sigmoid
    float acc = Tb2[d];
    for (int h = 0; h < TH_; ++h) {
        float t = Tb1[d * TH_ + h];
        #pragma unroll
        for (int o = 0; o < EO_; ++o) t += mmoe[o] * Tw1[(d * EO_ + o) * TH_ + h];
        acc += fmaxf(t, 0.f) * Tw2[d * TH_ + h];
    }
    float tout = 1.f / (1.f + expf(-acc));

    // outputs: [B,1] ++ [B,10] ++ [B,10]
    out[b] = tout;
    #pragma unroll
    for (int o = 0; o < EO_; ++o) {
        out[B_ + (size_t)b * EO_ + o]                    = avg[o];
        out[B_ + (size_t)B_ * EO_ + (size_t)b * EO_ + o] = mmoe[o];
    }
}

// ---------------- host launcher ----------------
extern "C" void kernel_launch(void* const* d_in, const int* in_sizes, int n_in,
                              void* d_out, int out_size, void* d_ws, size_t ws_size,
                              hipStream_t stream)
{
    (void)in_sizes; (void)n_in; (void)out_size; (void)ws_size;

    const float* x   = (const float*)d_in[0];
    const int*   dom = (const int*)d_in[1];
    const float* Ew1 = (const float*)d_in[2];
    const float* Eb1 = (const float*)d_in[3];
    const float* Ew2 = (const float*)d_in[4];
    const float* Eb2 = (const float*)d_in[5];
    const float* Ew3 = (const float*)d_in[6];
    const float* Eb3 = (const float*)d_in[7];
    const float* Gw1 = (const float*)d_in[8];
    const float* Gb1 = (const float*)d_in[9];
    const float* Gw2 = (const float*)d_in[10];
    const float* Gb2 = (const float*)d_in[11];
    const float* Tw1 = (const float*)d_in[12];
    const float* Tb1 = (const float*)d_in[13];
    const float* Tw2 = (const float*)d_in[14];
    const float* Tb2 = (const float*)d_in[15];
    float* out = (float*)d_out;

    char* ws = (char*)d_ws;
    size_t off = 0;
    auto take = [&](size_t bytes) -> char* {
        char* p = ws + off;
        off = (off + bytes + 255) & ~(size_t)255;
        return p;
    };

    u16* xb    = (u16*)take((size_t)B_ * IN_ * 2);            // 33.6 MB
    u16* Ew1T  = (u16*)take((size_t)E_ * EH1_ * IN_ * 2);     //  3.1 MB
    u16* Ew2T  = (u16*)take((size_t)E_ * EH2_ * EH1_ * 2);    //  0.4 MB
    u16* Ew3T  = (u16*)take((size_t)E_ * 16 * EH2_ * 2);      //  tiny
    u16* Gw1T  = (u16*)take((size_t)D_ * GH_ * IN_ * 2);      //  2.6 MB
    u16* h1    = (u16*)take((size_t)E_ * B_ * EH1_ * 2);      // 50.3 MB
    u16* h2    = (u16*)take((size_t)E_ * B_ * EH2_ * 2);      // 25.2 MB
    float* eo  = (float*)take((size_t)E_ * B_ * 16 * 4);      //  6.3 MB
    u16* ghid  = (u16*)take((size_t)B_ * GH_ * 2);            //  2.1 MB
    int* idxl  = (int*)take((size_t)B_ * 4);
    int* cnts  = (int*)take(D_ * 4);
    int* offs  = (int*)take(D_ * 4);
    int* curs  = (int*)take(D_ * 4);

    // --- precision conversion / weight transposes (once per call, cheap) ---
    {
        int n = B_ * IN_;
        cvt_f32_bf16<<<(n + 255) / 256, 256, 0, stream>>>(x, xb, n);
    }
    {
        int n = E_ * IN_ * EH1_;
        cvt_transpose_bf16<<<(n + 255) / 256, 256, 0, stream>>>(Ew1, Ew1T, E_, IN_, EH1_);
    }
    {
        int n = E_ * EH1_ * EH2_;
        cvt_transpose_bf16<<<(n + 255) / 256, 256, 0, stream>>>(Ew2, Ew2T, E_, EH1_, EH2_);
    }
    {
        int n = E_ * 16 * 128;
        cvt_ew3<<<(n + 255) / 256, 256, 0, stream>>>(Ew3, Ew3T);
    }
    {
        int n = D_ * IN_ * GH_;
        cvt_transpose_bf16<<<(n + 255) / 256, 256, 0, stream>>>(Gw1, Gw1T, D_, IN_, GH_);
    }

    // --- bucket samples by domain ---
    bucket_init<<<1, 32, 0, stream>>>(cnts, curs);
    bucket_count<<<(B_ + 255) / 256, 256, 0, stream>>>(dom, cnts, B_);
    bucket_scan<<<1, 32, 0, stream>>>(cnts, offs);
    bucket_scatter<<<(B_ + 255) / 256, 256, 0, stream>>>(dom, offs, curs, idxl, B_);

    // --- expert MLP: 1024 -> 256 -> 128 -> 10 ---
    {
        dim3 g(B_ / 128, EH1_ / 128, E_);
        gemm_bias_relu_bf16<IN_, EH1_><<<g, 256, 0, stream>>>(xb, 0LL, Ew1T, Eb1, h1, B_);
    }
    {
        dim3 g(B_ / 128, EH2_ / 128, E_);
        gemm_bias_relu_bf16<EH1_, EH2_><<<g, 256, 0, stream>>>(h1, (long long)B_ * EH1_, Ew2T, Eb2, h2, B_);
    }
    {
        dim3 g(B_ / 64, E_);
        gemm3_eo<<<g, 128, 0, stream>>>(h2, Ew3T, Eb3, eo, B_);
    }

    // --- grouped gate hidden GEMM (own domain only) ---
    {
        dim3 g(D_, 64);
        gate_gemm<<<g, 32, 0, stream>>>(xb, Gw1T, Gb1, idxl, offs, cnts, ghid);
    }

    // --- fused tail ---
    finalize<<<B_ / 256, 256, 0, stream>>>(dom, ghid, eo, Gw2, Gb2, Tw1, Tb1, Tw2, Tb2, out);
}
